// MultiHeadAttention_19902878450038
// MI455X (gfx1250) — matrix operand order repaired and checked by reading, not compiled, not run on hardware
//
#include <hip/hip_runtime.h>
#include <hip/hip_bf16.h>
#include <stdint.h>

#define HEADS  16
#define DK     64
#define DV     64
#define DMODEL 1024
#define NBATCH 2
#define TSEQ   2048
#define NT     (NBATCH * TSEQ)   // 4096 total rows
#define NKT    (DMODEL / 32)     // 32 K-tiles per GEMM

typedef __attribute__((ext_vector_type(16))) __bf16 bf16x16;
typedef __attribute__((ext_vector_type(8)))  float  f32x8;

union AB { bf16x16 v; uint4 q[2]; };

__device__ __forceinline__ f32x8 wmma_bf16(const AB& a, const AB& b, f32x8 c) {
    return __builtin_amdgcn_wmma_f32_16x16x32_bf16(false, a.v, false, b.v,
                                                   (short)0, c, false, false);
}

__device__ __forceinline__ unsigned short f2bf(float x) {
    union { float f; uint32_t u; } v; v.f = x;
    uint32_t r = v.u + 0x7FFFu + ((v.u >> 16) & 1u);   // round-to-nearest-even
    return (unsigned short)(r >> 16);
}

// ---------------------------------------------------------------------------
// Async global->LDS staging (CDNA5 GLOBAL_LOAD_ASYNC_TO_LDS_B128, ASYNCcnt).
// Toolchain signature: (int4 addrspace(1)*, int4 addrspace(3)*, imm, imm).
// Falls back to a VGPR round-trip copy if the builtin is not declared.
// ---------------------------------------------------------------------------
#if defined(__AMDGCN__) && __has_builtin(__builtin_amdgcn_global_load_async_to_lds_b128)
#define HAVE_ASYNC_LDS 1
#endif

#ifdef HAVE_ASYNC_LDS
typedef int v4i_ __attribute__((vector_size(4 * sizeof(int))));
typedef __attribute__((address_space(1))) v4i_* gas_v4i_ptr;
typedef __attribute__((address_space(3))) v4i_* las_v4i_ptr;
#endif

__device__ __forceinline__ void stage16_async(const unsigned short* g, unsigned short* l) {
#ifdef HAVE_ASYNC_LDS
    __builtin_amdgcn_global_load_async_to_lds_b128(
        (gas_v4i_ptr)const_cast<unsigned short*>(g),
        (las_v4i_ptr)l, 0, 0);
#else
    *(uint4*)l = *(const uint4*)g;
#endif
}

__device__ __forceinline__ void stage_wait() {
#ifdef HAVE_ASYNC_LDS
    __asm__ volatile("s_wait_asynccnt 0" ::: "memory");
#endif
}

// ---------------------------------------------------------------------------
// Kernel 1: transpose + convert all four weight matrices to bf16 (N,K) major.
// ---------------------------------------------------------------------------
__global__ __launch_bounds__(256) void pack_w_kernel(
    const float* __restrict__ Wq, const float* __restrict__ Wk,
    const float* __restrict__ Wv, const float* __restrict__ Wo,
    unsigned short* __restrict__ ws) {
    const float* srcs[4] = {Wq, Wk, Wv, Wo};
    const float* src = srcs[blockIdx.y];
    unsigned short* dst = ws + (size_t)blockIdx.y * (DMODEL * DMODEL);
    int lin = blockIdx.x * 256 + threadIdx.x;
    int k = lin / DMODEL, n = lin % DMODEL;
    dst[(size_t)n * DMODEL + k] = f2bf(src[(size_t)k * DMODEL + n]);
}

// ---------------------------------------------------------------------------
// Kernel 2: projection GEMM  Y = X(fp32) @ W, output bf16 in head-split layout.
// mode 0: dst[n][h][t][d]  (Q and K)     mode 1: dst[n][h][d][t]  (V, transposed)
// Double-buffered LDS: async DMA of tile k+1 overlaps WMMA on tile k.
// ---------------------------------------------------------------------------
__global__ __launch_bounds__(256) void proj_gemm_kernel(
    const float* __restrict__ X, const unsigned short* __restrict__ Wt,
    unsigned short* __restrict__ dst, int mode) {
    __shared__ unsigned short As[2][128 * 32];   // rows m, k-stride 32
    __shared__ unsigned short Bs[2][128 * 32];   // rows n (out col), k-stride 32
    const int tid  = threadIdx.x;
    const int lane = tid & 31, wid = tid >> 5;
    const int wm = wid >> 1, wn = wid & 1;
    const int lm = lane & 15, lh = lane >> 4;
    const int rowBase = blockIdx.y * 128;
    const int colBase = blockIdx.x * 128;

    f32x8 acc[2][4];
    for (int i = 0; i < 2; i++) for (int j = 0; j < 4; j++)
        for (int e = 0; e < 8; e++) acc[i][j][e] = 0.f;

    auto stage_tiles = [&](int buf, int k0) {
        // B tile via async DMA (overlaps the A conversion + the caller's WMMAs)
        for (int i = 0; i < 2; i++) {
            int lin = tid + i * 256;                 // 512 chunks of 8 bf16
            int r = lin >> 2, kc = (lin & 3) << 3;
            stage16_async(&Wt[(size_t)(colBase + r) * DMODEL + k0 + kc],
                          &Bs[buf][r * 32 + kc]);
        }
        // A tile: 128x32 fp32 -> bf16 through VGPRs
        for (int i = 0; i < 4; i++) {
            int lin = tid + i * 256;                 // 1024 float4 chunks
            int r = lin >> 3, kc = (lin & 7) << 2;
            float4 f = *(const float4*)&X[(size_t)(rowBase + r) * DMODEL + k0 + kc];
            ushort4 u; u.x = f2bf(f.x); u.y = f2bf(f.y); u.z = f2bf(f.z); u.w = f2bf(f.w);
            *(ushort4*)&As[buf][r * 32 + kc] = u;
        }
    };

    stage_tiles(0, 0);
    for (int kt = 0; kt < NKT; kt++) {
        const int cur = kt & 1;
        stage_wait();          // our async writes into buf[cur] done
        __syncthreads();       // everyone's DMA done; buf[cur^1] fully read
        if (kt + 1 < NKT) stage_tiles(cur ^ 1, (kt + 1) * 32);

        AB a[2], b[4];
        for (int mi = 0; mi < 2; mi++) {
            int m = wm * 32 + mi * 16 + lm;
            a[mi].q[0] = *(const uint4*)&As[cur][m * 32 + lh * 8];
            a[mi].q[1] = *(const uint4*)&As[cur][m * 32 + lh * 8 + 16];
        }
        for (int ni = 0; ni < 4; ni++) {
            int nr = wn * 64 + ni * 16 + lm;
            b[ni].q[0] = *(const uint4*)&Bs[cur][nr * 32 + lh * 8];
            b[ni].q[1] = *(const uint4*)&Bs[cur][nr * 32 + lh * 8 + 16];
        }
        for (int mi = 0; mi < 2; mi++)
            for (int ni = 0; ni < 4; ni++)
                acc[mi][ni] = wmma_bf16(a[mi], b[ni], acc[mi][ni]);
    }

    for (int mi = 0; mi < 2; mi++)
        for (int ni = 0; ni < 4; ni++)
            for (int r = 0; r < 8; r++) {
                int row = rowBase + wm * 32 + mi * 16 + r + lh * 8;
                int col = colBase + wn * 64 + ni * 16 + lm;
                int nb = row / TSEQ, t = row % TSEQ;
                int h = col / DK, d = col % DK;
                unsigned short v = f2bf(acc[mi][ni][r]);
                if (mode == 0)
                    dst[(((size_t)(nb * HEADS + h) * TSEQ) + t) * DK + d] = v;
                else
                    dst[(((size_t)(nb * HEADS + h) * DK) + d) * TSEQ + t] = v;
            }
}

// ---------------------------------------------------------------------------
// Kernel 3: flash-attention. One wave = 16 query rows; block = 4 waves = 64 q.
// grid = (T/64, HEADS, N). Streams keys in chunks of 32 with online softmax.
// P reshape uses per-wave-private LDS (wave LDS ops are in-order -> no barrier).
// ---------------------------------------------------------------------------
__global__ __launch_bounds__(128) void attn_kernel(
    const unsigned short* __restrict__ Qb, const unsigned short* __restrict__ Kb,
    const unsigned short* __restrict__ Vt, const int* __restrict__ mask,
    unsigned short* __restrict__ concat) {
    __shared__ unsigned short Pt[4][16][32];   // per-wave P tile (bf16)
    const int lane = threadIdx.x & 31, wid = threadIdx.x >> 5;
    const int lm = lane & 15, lh = lane >> 4;
    const int h = blockIdx.y, n = blockIdx.z;
    const int q0 = blockIdx.x * 64 + wid * 16;

    const size_t headQK = (size_t)(n * HEADS + h) * TSEQ * DK;

    // Q A-fragments (16 rows x 64 d) kept in registers for the whole loop
    AB qf[2];
    {
        const unsigned short* qrow = Qb + headQK + (size_t)(q0 + lm) * DK;
        for (int ks = 0; ks < 2; ks++) {
            qf[ks].q[0] = *(const uint4*)&qrow[ks * 32 + lh * 8];
            qf[ks].q[1] = *(const uint4*)&qrow[ks * 32 + lh * 8 + 16];
        }
    }

    f32x8 o[4];
    for (int d = 0; d < 4; d++) for (int e = 0; e < 8; e++) o[d][e] = 0.f;
    float mrun[8], lrun[8];
    for (int r = 0; r < 8; r++) { mrun[r] = -3.0e38f; lrun[r] = 0.f; }

    const int* mrow = mask + ((size_t)n * TSEQ + q0 + lh * 8) * TSEQ;
    const unsigned short* vhead = Vt + ((size_t)(n * HEADS + h) * DV + lm) * TSEQ;

#pragma unroll 2
    for (int kb = 0; kb < TSEQ; kb += 32) {
        // prefetch next chunk's K row and V row toward the WGP caches
        if (kb + 32 < TSEQ) {
            __builtin_prefetch(Kb + headQK + (size_t)(kb + 32 + lm) * DK, 0, 3);
            __builtin_prefetch(vhead + kb + 32 + lh * 16, 0, 3);
        }
        // S = Q @ K^T for two 16-key blocks (K-dim = 64 -> two k-steps)
        f32x8 s[2];
        for (int blk = 0; blk < 2; blk++) {
            for (int e = 0; e < 8; e++) s[blk][e] = 0.f;
            const unsigned short* krow = Kb + headQK + (size_t)(kb + blk * 16 + lm) * DK;
            for (int ks = 0; ks < 2; ks++) {
                AB bf;
                bf.q[0] = *(const uint4*)&krow[ks * 32 + lh * 8];
                bf.q[1] = *(const uint4*)&krow[ks * 32 + lh * 8 + 16];
                s[blk] = wmma_bf16(qf[ks], bf, s[blk]);
            }
        }
        // scale + mask (reference: mask==0 -> -1e20)
        for (int blk = 0; blk < 2; blk++) {
            int key = kb + blk * 16 + lm;
            for (int r = 0; r < 8; r++) {
                float v = s[blk][r] * 0.125f;            // 1/sqrt(64)
                if (mrow[(size_t)r * TSEQ + key] == 0) v = -1.0e20f;
                s[blk][r] = v;
            }
        }
        // online softmax: rows live in lanes 0-15 / 16-31 -> 16-wide butterflies
        float pf[2][8];
        for (int r = 0; r < 8; r++) {
            float rowm = fmaxf(s[0][r], s[1][r]);
            for (int off = 8; off >= 1; off >>= 1)
                rowm = fmaxf(rowm, __shfl_xor(rowm, off, 16));
            float mnew  = fmaxf(mrun[r], rowm);
            float alpha = __expf(mrun[r] - mnew);
            float p0 = __expf(s[0][r] - mnew);
            float p1 = __expf(s[1][r] - mnew);
            float rs = p0 + p1;
            for (int off = 8; off >= 1; off >>= 1)
                rs += __shfl_xor(rs, off, 16);
            lrun[r] = lrun[r] * alpha + rs;
            mrun[r] = mnew;
            for (int d = 0; d < 4; d++) o[d][r] *= alpha;   // same row mapping as S
            pf[0][r] = p0; pf[1][r] = p1;
        }
        // C-layout -> A-fragment reshape of P via per-wave LDS region.
        // Wave LDS ops are in-order (DScnt); only stop compiler reordering.
        for (int blk = 0; blk < 2; blk++)
            for (int r = 0; r < 8; r++)
                Pt[wid][r + lh * 8][blk * 16 + lm] = f2bf(pf[blk][r]);
        __asm__ volatile("" ::: "memory");
        AB pfr;
        pfr.q[0] = *(const uint4*)&Pt[wid][lm][lh * 16];
        pfr.q[1] = *(const uint4*)&Pt[wid][lm][lh * 16 + 8];
        __asm__ volatile("" ::: "memory");
        // O += P @ V   (V pre-transposed: rows = d, keys contiguous)
        const unsigned short* vbase = vhead + kb + lh * 16;
        for (int d = 0; d < 4; d++) {
            AB vf;
            const unsigned short* vrow = vbase + (size_t)d * 16 * TSEQ;
            vf.q[0] = *(const uint4*)&vrow[0];
            vf.q[1] = *(const uint4*)&vrow[8];
            o[d] = wmma_bf16(pfr, vf, o[d]);
        }
    }

    // normalize and store concat (n, t, h*64+d) in bf16 for the output GEMM
    for (int d = 0; d < 4; d++)
        for (int r = 0; r < 8; r++) {
            float v = o[d][r] / lrun[r];
            int t = q0 + r + lh * 8;
            int col = h * DV + d * 16 + lm;
            concat[((size_t)n * TSEQ + t) * DMODEL + col] = f2bf(v);
        }
}

// ---------------------------------------------------------------------------
// Kernel 4: output GEMM  out(fp32) = concat(bf16) @ Wo  (Wo pre-transposed bf16)
// Both tiles async-DMA staged, double-buffered.
// ---------------------------------------------------------------------------
__global__ __launch_bounds__(256) void out_gemm_kernel(
    const unsigned short* __restrict__ Xb, const unsigned short* __restrict__ Wt,
    float* __restrict__ out) {
    __shared__ unsigned short As[2][128 * 32];
    __shared__ unsigned short Bs[2][128 * 32];
    const int tid  = threadIdx.x;
    const int lane = tid & 31, wid = tid >> 5;
    const int wm = wid >> 1, wn = wid & 1;
    const int lm = lane & 15, lh = lane >> 4;
    const int rowBase = blockIdx.y * 128;
    const int colBase = blockIdx.x * 128;

    f32x8 acc[2][4];
    for (int i = 0; i < 2; i++) for (int j = 0; j < 4; j++)
        for (int e = 0; e < 8; e++) acc[i][j][e] = 0.f;

    auto stage_tiles = [&](int buf, int k0) {
        for (int i = 0; i < 2; i++) {
            int lin = tid + i * 256;
            int r = lin >> 2, kc = (lin & 3) << 3;
            stage16_async(&Xb[(size_t)(rowBase + r) * DMODEL + k0 + kc],
                          &As[buf][r * 32 + kc]);
            stage16_async(&Wt[(size_t)(colBase + r) * DMODEL + k0 + kc],
                          &Bs[buf][r * 32 + kc]);
        }
    };

    stage_tiles(0, 0);
    for (int kt = 0; kt < NKT; kt++) {
        const int cur = kt & 1;
        stage_wait();
        __syncthreads();
        if (kt + 1 < NKT) stage_tiles(cur ^ 1, (kt + 1) * 32);

        AB a[2], b[4];
        for (int mi = 0; mi < 2; mi++) {
            int m = wm * 32 + mi * 16 + lm;
            a[mi].q[0] = *(const uint4*)&As[cur][m * 32 + lh * 8];
            a[mi].q[1] = *(const uint4*)&As[cur][m * 32 + lh * 8 + 16];
        }
        for (int ni = 0; ni < 4; ni++) {
            int nr = wn * 64 + ni * 16 + lm;
            b[ni].q[0] = *(const uint4*)&Bs[cur][nr * 32 + lh * 8];
            b[ni].q[1] = *(const uint4*)&Bs[cur][nr * 32 + lh * 8 + 16];
        }
        for (int mi = 0; mi < 2; mi++)
            for (int ni = 0; ni < 4; ni++)
                acc[mi][ni] = wmma_bf16(a[mi], b[ni], acc[mi][ni]);
    }

    for (int mi = 0; mi < 2; mi++)
        for (int ni = 0; ni < 4; ni++)
            for (int r = 0; r < 8; r++) {
                int row = rowBase + wm * 32 + mi * 16 + r + lh * 8;
                int col = colBase + wn * 64 + ni * 16 + lm;
                out[(size_t)row * DMODEL + col] = acc[mi][ni][r];
            }
}

// ---------------------------------------------------------------------------
extern "C" void kernel_launch(void* const* d_in, const int* in_sizes, int n_in,
                              void* d_out, int out_size, void* d_ws, size_t ws_size,
                              hipStream_t stream) {
    (void)in_sizes; (void)n_in; (void)out_size; (void)ws_size;
    const float* q    = (const float*)d_in[0];
    const float* k    = (const float*)d_in[1];
    const float* v    = (const float*)d_in[2];
    const int*   mask = (const int*)  d_in[3];
    const float* Wq   = (const float*)d_in[4];
    const float* Wk   = (const float*)d_in[5];
    const float* Wv   = (const float*)d_in[6];
    const float* Wo   = (const float*)d_in[7];
    float* out = (float*)d_out;

    // workspace layout (bf16 elements): 4x1M weights + 3x4M activations + 4M concat = 40MB
    unsigned short* ws  = (unsigned short*)d_ws;
    unsigned short* WqT = ws;
    unsigned short* WkT = WqT + (size_t)DMODEL * DMODEL;
    unsigned short* WvT = WkT + (size_t)DMODEL * DMODEL;
    unsigned short* WoT = WvT + (size_t)DMODEL * DMODEL;
    unsigned short* Qb  = WoT + (size_t)DMODEL * DMODEL;
    unsigned short* Kb  = Qb  + (size_t)NT * DMODEL;
    unsigned short* Vt  = Kb  + (size_t)NT * DMODEL;
    unsigned short* Cc  = Vt  + (size_t)NT * DMODEL;

    pack_w_kernel<<<dim3(DMODEL * DMODEL / 256, 4), 256, 0, stream>>>(Wq, Wk, Wv, Wo, ws);

    dim3 g(DMODEL / 128, NT / 128);
    proj_gemm_kernel<<<g, 256, 0, stream>>>(q, WqT, Qb, 0);
    proj_gemm_kernel<<<g, 256, 0, stream>>>(k, WkT, Kb, 0);
    proj_gemm_kernel<<<g, 256, 0, stream>>>(v, WvT, Vt, 1);

    attn_kernel<<<dim3(TSEQ / 64, HEADS, NBATCH), 128, 0, stream>>>(Qb, Kb, Vt, mask, Cc);

    out_gemm_kernel<<<g, 256, 0, stream>>>(Cc, WoT, out);
}
